// DualChannelGCN_36275293782543
// MI455X (gfx1250) — matrix operand order repaired and checked by reading, not compiled, not run on hardware
//
#include <hip/hip_runtime.h>

#define BB 64
#define NN 256
#define DD 768
#define GG 300
#define GP 304      // G padded to multiple of 16 (N dim of GEMMs)
#define KP1 320     // G padded to multiple of 32 (K dim of layer-1 GEMMs)
#define KPF 640     // 2G padded to multiple of 32 (K dim of fusion GEMM)
#define ROWS (BB*NN)   // 16384

typedef _Float16 half8 __attribute__((ext_vector_type(8)));
typedef _Float16 v16h  __attribute__((ext_vector_type(16)));
typedef float    v8f   __attribute__((ext_vector_type(8)));

// ---------------------------------------------------------------------------
// f32 -> f16 with column zero-padding: src [rows, scols] -> dst [rows, dcols]
__global__ void k_f32_to_f16_pad(const float* __restrict__ src, _Float16* __restrict__ dst,
                                 int scols, int dcols, long total) {
    long i = (long)blockIdx.x * 256 + threadIdx.x;
    if (i >= total) return;
    long r = i / dcols; int c = (int)(i % dcols);
    dst[i] = (c < scols) ? (_Float16)src[r * (long)scols + c] : (_Float16)0.0f;
}

// Weight convert+transpose: src f32 [K, G] row-major -> dst f16 col-major [GP, KP]
__global__ void k_weightT(const float* __restrict__ src, _Float16* __restrict__ dst,
                          int K, int G, int KPd) {
    long i = (long)blockIdx.x * 256 + threadIdx.x;
    long total = (long)GP * KPd;
    if (i >= total) return;
    int n = (int)(i / KPd); int k = (int)(i % KPd);
    dst[i] = (n < G && k < K) ? (_Float16)src[(long)k * G + n] : (_Float16)0.0f;
}

// ---------------------------------------------------------------------------
// Register-blocked batched WMMA GEMM: C[M,Nn] = A[M,K] (f16 row-major, lda) *
// Bt (f16 COLUMN-major: Bt[n][k], ldb = stride per column = K).
// 4 waves/block; each wave owns a 64(M) x 16(N) strip: 4 f32 accumulators,
// B fragment loaded once per K-step and reused by 4 v_wmma_f32_16x16x32_f16.
__global__ void k_gemm(const _Float16* __restrict__ A, const _Float16* __restrict__ Bt,
                       float* __restrict__ C, int M, int Nn, int K,
                       int lda, int ldb, int ldc,
                       long strideA, long strideB, long strideC) {
    int bz = blockIdx.z;
    A  += (long)bz * strideA;
    Bt += (long)bz * strideB;
    C  += (long)bz * strideC;
    int wave = threadIdx.x >> 5;
    int lane = threadIdx.x & 31;
    int m0 = (blockIdx.y * 4 + wave) * 64;     // 64 rows per wave
    int n0 = blockIdx.x * 16;
    if (m0 >= M) return;
    int nc = n0 + (lane & 15);                 // Bt column this lane reads
    int hi = lane >> 4;                        // lane-half select
    v8f acc[4];
    #pragma unroll
    for (int t = 0; t < 4; t++) acc[t] = (v8f){0.f,0.f,0.f,0.f,0.f,0.f,0.f,0.f};
    // A lane base: row m0+(lane&15), chunk offset 8*hi within each 32-wide k block
    const _Float16* arow = A + (long)(m0 + (lane & 15)) * lda + 8 * hi;
    const _Float16* bcol = Bt + (long)nc * ldb + 8 * hi;
    long mstep = (long)16 * lda;               // 16 output rows
    for (int k0 = 0; k0 < K; k0 += 32) {
        // B lane layout: halves 0..7 -> k = k0+8*hi+(0..7); 8..15 -> +16, fixed n (shared by 4 WMMAs)
        const half8* pb = (const half8*)(bcol + k0);
        half8 b0 = pb[0];
        half8 b1 = pb[2];
        v16h bv;
        #pragma unroll
        for (int i = 0; i < 8; i++) { bv[i] = b0[i]; bv[i + 8] = b1[i]; }
        #pragma unroll
        for (int t = 0; t < 4; t++) {
            // A lane layout: halves 0..7 -> k = k0+8*hi+(0..7); 8..15 -> +16
            const half8* pa = (const half8*)(arow + (long)t * mstep + k0);
            half8 a0 = pa[0];
            half8 a1 = pa[2];                  // +16 halves
            v16h av;
            #pragma unroll
            for (int i = 0; i < 8; i++) { av[i] = a0[i]; av[i + 8] = a1[i]; }
            acc[t] = __builtin_amdgcn_wmma_f32_16x16x32_f16(false, av, false, bv,
                                                            (short)0, acc[t], false, false);
        }
    }
    // C layout: vgpr r -> row (tile base) + r + 8*hi, col n0 + (lane&15)
    #pragma unroll
    for (int t = 0; t < 4; t++) {
        float* crow = C + (long)(m0 + t * 16 + 8 * hi) * ldc + n0 + (lane & 15);
        #pragma unroll
        for (int r = 0; r < 8; r++) crow[(long)r * ldc] = acc[t][r];
    }
}

// ---------------------------------------------------------------------------
// Wh += pos_emb[positions]; f1 = Wh.a1; f2 = Wh.a2   (one wave per row)
__global__ void k_pos_f1f2(float* __restrict__ Wh, const float* __restrict__ pos,
                           const int* __restrict__ positions,
                           const float* __restrict__ a1, const float* __restrict__ a2,
                           float* __restrict__ f1, float* __restrict__ f2) {
    int row  = blockIdx.x * 8 + (threadIdx.x >> 5);
    int lane = threadIdx.x & 31;
    int p = positions[row];
    float* w = Wh + (long)row * GP;
    const float* pr = pos + (long)p * GG;
    float s1 = 0.f, s2 = 0.f;
    for (int g = lane; g < GG; g += 32) {
        float v = w[g] + pr[g];
        w[g] = v;
        s1 += v * a1[g];
        s2 += v * a2[g];
    }
    for (int off = 16; off; off >>= 1) {
        s1 += __shfl_xor(s1, off, 32);
        s2 += __shfl_xor(s2, off, 32);
    }
    if (lane == 0) { f1[row] = s1; f2[row] = s2; }
}

// ---------------------------------------------------------------------------
// Masked-softmax attention row: one wave per (b,i). adjI!=null -> syntactic
// int mask; else semantic mask 0.5*(s[i]+s[j]) + (i==j) > 0. Writes f16 attn.
__global__ void k_attn(const float* __restrict__ f1, const float* __restrict__ f2,
                       const int* __restrict__ adjI, const float* __restrict__ s,
                       _Float16* __restrict__ attn) {
    int row  = blockIdx.x * 8 + (threadIdx.x >> 5);
    int lane = threadIdx.x & 31;
    int b = row >> 8, i = row & (NN - 1);
    float fi = f1[row];
    float si = adjI ? 0.f : s[(long)b * NN + i];
    float ev[8];
    float mx = -1e30f;
    #pragma unroll
    for (int t = 0; t < 8; t++) {
        int j = lane + t * 32;
        float e = fi + f2[(long)b * NN + j];
        e = (e > 0.f) ? e : 0.2f * e;                    // leaky_relu(0.2)
        bool keep;
        if (adjI) keep = adjI[(long)row * NN + j] > 0;
        else {
            float av = 0.5f * (si + s[(long)b * NN + j]) + ((i == j) ? 1.f : 0.f);
            keep = av > 0.f;
        }
        e = keep ? e : -9e15f;
        ev[t] = e;
        mx = fmaxf(mx, e);
    }
    for (int off = 16; off; off >>= 1) mx = fmaxf(mx, __shfl_xor(mx, off, 32));
    float sum = 0.f;
    #pragma unroll
    for (int t = 0; t < 8; t++) { ev[t] = __expf(ev[t] - mx); sum += ev[t]; }
    for (int off = 16; off; off >>= 1) sum += __shfl_xor(sum, off, 32);
    float inv = 1.f / sum;
    #pragma unroll
    for (int t = 0; t < 8; t++)
        attn[(long)row * NN + lane + t * 32] = (_Float16)(ev[t] * inv);
}

// Wh f32 [B*N, GP] -> WhT f16 [B, GP, N] (column-major per batch for GEMM B)
__global__ void k_transpose16(const float* __restrict__ Wh, _Float16* __restrict__ whT) {
    int g = blockIdx.x, b = blockIdx.y, n = threadIdx.x;
    whT[((long)b * GP + g) * NN + n] = (_Float16)Wh[((long)b * NN + n) * GP + g];
}

// ---------------------------------------------------------------------------
// y = relu(layernorm(hp + res (+tb)) * g + b). Writes compact f32 (optional)
// and padded f16 at column offset (for the next GEMM / fusion concat).
__global__ void k_ln(const float* __restrict__ hp, const float* __restrict__ res, int res_ld,
                     const float* __restrict__ tb,
                     const float* __restrict__ gw, const float* __restrict__ gb,
                     float* __restrict__ out_c, _Float16* __restrict__ out16,
                     int ld16, int coloff) {
    int row  = blockIdx.x * 8 + (threadIdx.x >> 5);
    int lane = threadIdx.x & 31;
    const float* hr = hp  + (long)row * GP;
    const float* rr = res + (long)row * res_ld;
    float xv[10];
    float s = 0.f, ss = 0.f;
    int cnt = 0;
    for (int g = lane; g < GG; g += 32) {
        float x = hr[g] + rr[g] + (tb ? tb[g] : 0.f);
        xv[cnt++] = x;
        s += x; ss += x * x;
    }
    for (int off = 16; off; off >>= 1) {
        s  += __shfl_xor(s,  off, 32);
        ss += __shfl_xor(ss, off, 32);
    }
    float mean = s * (1.f / GG);
    float var  = ss * (1.f / GG) - mean * mean;
    float rstd = rsqrtf(var + 1e-5f);
    cnt = 0;
    for (int g = lane; g < GG; g += 32) {
        float y = (xv[cnt++] - mean) * rstd * gw[g] + gb[g];
        y = fmaxf(y, 0.f);
        if (out_c) out_c[(long)row * GG + g] = y;
        out16[(long)row * ld16 + coloff + g] = (_Float16)y;
    }
}

// Fusion epilogue: out = relu(C + bias), compact to [B*N, G]
__global__ void k_bias_relu(const float* __restrict__ C, const float* __restrict__ bias,
                            float* __restrict__ out) {
    long i = (long)blockIdx.x * 256 + threadIdx.x;
    if (i >= (long)ROWS * GG) return;
    long r = i / GG; int g = (int)(i % GG);
    out[i] = fmaxf(C[r * GP + g] + bias[g], 0.f);
}

// ---------------------------------------------------------------------------
// Semantic prep: masked aspect mean + inverse norm per batch
__global__ void k_aspect(const float* __restrict__ h, const float* __restrict__ mask,
                         float* __restrict__ arepr, float* __restrict__ ainv) {
    int b = blockIdx.x, t = threadIdx.x;
    __shared__ float sm[NN];
    __shared__ float red[NN];
    sm[t] = mask[(long)b * NN + t];
    __syncthreads();
    red[t] = sm[t];
    __syncthreads();
    for (int off = 128; off; off >>= 1) { if (t < off) red[t] += red[t + off]; __syncthreads(); }
    float inv = 1.f / (red[0] + 1e-8f);
    __syncthreads();
    float loc[3];
    #pragma unroll
    for (int c = 0; c < 3; c++) {
        int d = t + c * NN;
        float acc = 0.f;
        for (int n = 0; n < NN; n++) acc += h[((long)b * NN + n) * DD + d] * sm[n];
        float v = acc * inv;
        arepr[(long)b * DD + d] = v;
        loc[c] = v;
    }
    red[t] = loc[0]*loc[0] + loc[1]*loc[1] + loc[2]*loc[2];
    __syncthreads();
    for (int off = 128; off; off >>= 1) { if (t < off) red[t] += red[t + off]; __syncthreads(); }
    if (t == 0) ainv[b] = 1.f / fmaxf(sqrtf(red[0]), 1e-12f);
}

// Cosine similarity to aspect repr: one wave per (b,n)
__global__ void k_sim(const float* __restrict__ h, const float* __restrict__ arepr,
                      const float* __restrict__ ainv, float* __restrict__ sim) {
    int row  = blockIdx.x * 8 + (threadIdx.x >> 5);
    int lane = threadIdx.x & 31;
    int b = row >> 8;
    const float* hr = h + (long)row * DD;
    const float* ar = arepr + (long)b * DD;
    float d = 0.f, hh = 0.f;
    for (int k = lane; k < DD; k += 32) { float x = hr[k]; d += x * ar[k]; hh += x * x; }
    for (int off = 16; off; off >>= 1) {
        d  += __shfl_xor(d,  off, 32);
        hh += __shfl_xor(hh, off, 32);
    }
    if (lane == 0) sim[row] = d * ainv[b] / fmaxf(sqrtf(hh), 1e-12f);
}

// Top-10 per batch -> scatter values into s[b,n] (s pre-zeroed)
__global__ void k_topk(const float* __restrict__ sim, float* __restrict__ s) {
    int b = blockIdx.x, t = threadIdx.x;
    __shared__ float vals[NN];
    __shared__ float v2[NN];
    __shared__ int   i2[NN];
    vals[t] = sim[(long)b * NN + t];
    __syncthreads();
    for (int it = 0; it < 10; it++) {
        v2[t] = vals[t]; i2[t] = t;
        __syncthreads();
        for (int off = 128; off; off >>= 1) {
            if (t < off && v2[t + off] > v2[t]) { v2[t] = v2[t + off]; i2[t] = i2[t + off]; }
            __syncthreads();
        }
        if (t == 0) { s[(long)b * NN + i2[0]] = v2[0]; vals[i2[0]] = -3.0e38f; }
        __syncthreads();
    }
}

// ---------------------------------------------------------------------------
static void run_layer(const _Float16* A16, int K,
                      const _Float16* Wt, const _Float16* tWt,
                      const float* a1, const float* a2, const float* posemb,
                      const float* tb, const float* gw, const float* gbias,
                      const int* positions, const int* adjI, const float* s,
                      const float* resc,
                      float* Wh, float* res, float* hp,
                      _Float16* attn16, _Float16* whT,
                      float* f1, float* f2,
                      float* out_c, _Float16* out16, int ld16, int coloff,
                      hipStream_t stream) {
    dim3 gbig(GP / 16, ROWS / 256, 1);         // 4 waves x 64 rows = 256 rows/block
    k_gemm<<<gbig, 128, 0, stream>>>(A16, Wt, Wh, ROWS, GP, K, K, K, GP, 0, 0, 0);
    if (tWt)
        k_gemm<<<gbig, 128, 0, stream>>>(A16, tWt, res, ROWS, GP, K, K, K, GP, 0, 0, 0);
    k_pos_f1f2<<<ROWS / 8, 256, 0, stream>>>(Wh, posemb, positions, a1, a2, f1, f2);
    k_attn<<<ROWS / 8, 256, 0, stream>>>(f1, f2, adjI, s, attn16);
    k_transpose16<<<dim3(GP, BB), NN, 0, stream>>>(Wh, whT);
    k_gemm<<<dim3(GP / 16, 1, BB), 128, 0, stream>>>(
        attn16, whT, hp, NN, GP, NN, NN, NN, GP,
        (long)NN * NN, (long)GP * NN, (long)NN * GP);
    k_ln<<<ROWS / 8, 256, 0, stream>>>(hp, tWt ? res : resc, tWt ? GP : GG,
                                       tWt ? tb : nullptr, gw, gbias,
                                       out_c, out16, ld16, coloff);
}

extern "C" void kernel_launch(void* const* d_in, const int* in_sizes, int n_in,
                              void* d_out, int out_size, void* d_ws, size_t ws_size,
                              hipStream_t stream) {
    const float* h        = (const float*)d_in[0];
    const float* amask    = (const float*)d_in[1];
    const float* syn0_W   = (const float*)d_in[2];
    const float* syn0_a1  = (const float*)d_in[3];
    const float* syn0_a2  = (const float*)d_in[4];
    const float* syn0_pos = (const float*)d_in[5];
    const float* syn0_tW  = (const float*)d_in[6];
    const float* syn0_tb  = (const float*)d_in[7];
    const float* syn0_g   = (const float*)d_in[8];
    const float* syn0_b   = (const float*)d_in[9];
    const float* syn1_W   = (const float*)d_in[10];
    const float* syn1_a1  = (const float*)d_in[11];
    const float* syn1_a2  = (const float*)d_in[12];
    const float* syn1_pos = (const float*)d_in[13];
    const float* syn1_g   = (const float*)d_in[14];
    const float* syn1_b   = (const float*)d_in[15];
    const float* sem0_W   = (const float*)d_in[16];
    const float* sem0_a1  = (const float*)d_in[17];
    const float* sem0_a2  = (const float*)d_in[18];
    const float* sem0_pos = (const float*)d_in[19];
    const float* sem0_tW  = (const float*)d_in[20];
    const float* sem0_tb  = (const float*)d_in[21];
    const float* sem0_g   = (const float*)d_in[22];
    const float* sem0_b   = (const float*)d_in[23];
    const float* sem1_W   = (const float*)d_in[24];
    const float* sem1_a1  = (const float*)d_in[25];
    const float* sem1_a2  = (const float*)d_in[26];
    const float* sem1_pos = (const float*)d_in[27];
    const float* sem1_g   = (const float*)d_in[28];
    const float* sem1_b   = (const float*)d_in[29];
    const float* fus_W    = (const float*)d_in[30];
    const float* fus_b    = (const float*)d_in[31];
    const int*   syn_adj  = (const int*)d_in[32];
    const int*   positions= (const int*)d_in[33];
    float* out = (float*)d_out;

    // ---- workspace bump allocator (256B aligned) ----
    char* p = (char*)d_ws;
    auto alloc = [&](size_t bytes) -> void* {
        void* r = (void*)p;
        p += (bytes + 255) & ~(size_t)255;
        return r;
    };
    _Float16* h16      = (_Float16*)alloc((size_t)ROWS * DD * 2);
    _Float16* w_s0     = (_Float16*)alloc((size_t)GP * DD * 2);
    _Float16* w_s0t    = (_Float16*)alloc((size_t)GP * DD * 2);
    _Float16* w_m0     = (_Float16*)alloc((size_t)GP * DD * 2);
    _Float16* w_m0t    = (_Float16*)alloc((size_t)GP * DD * 2);
    _Float16* w_s1     = (_Float16*)alloc((size_t)GP * KP1 * 2);
    _Float16* w_m1     = (_Float16*)alloc((size_t)GP * KP1 * 2);
    _Float16* w_fus    = (_Float16*)alloc((size_t)GP * KPF * 2);
    _Float16* attn16   = (_Float16*)alloc((size_t)BB * NN * NN * 2);
    _Float16* whT      = (_Float16*)alloc((size_t)BB * GP * NN * 2);
    _Float16* hs0_16   = (_Float16*)alloc((size_t)ROWS * KP1 * 2);
    _Float16* hm0_16   = (_Float16*)alloc((size_t)ROWS * KP1 * 2);
    _Float16* fusA16   = (_Float16*)alloc((size_t)ROWS * KPF * 2);
    float* Wh    = (float*)alloc((size_t)ROWS * GP * 4);
    float* res   = (float*)alloc((size_t)ROWS * GP * 4);
    float* hp    = (float*)alloc((size_t)ROWS * GP * 4);
    float* hs0c  = (float*)alloc((size_t)ROWS * GG * 4);
    float* hm0c  = (float*)alloc((size_t)ROWS * GG * 4);
    float* f1    = (float*)alloc((size_t)ROWS * 4);
    float* f2    = (float*)alloc((size_t)ROWS * 4);
    float* sim   = (float*)alloc((size_t)ROWS * 4);
    float* sbuf  = (float*)alloc((size_t)ROWS * 4);
    float* arepr = (float*)alloc((size_t)BB * DD * 4);
    float* ainv  = (float*)alloc((size_t)BB * 4);

    // zero-fill buffers whose pad columns must be 0
    hipMemsetAsync(sbuf,   0, (size_t)ROWS * 4, stream);
    hipMemsetAsync(hs0_16, 0, (size_t)ROWS * KP1 * 2, stream);
    hipMemsetAsync(hm0_16, 0, (size_t)ROWS * KP1 * 2, stream);
    hipMemsetAsync(fusA16, 0, (size_t)ROWS * KPF * 2, stream);

    // ---- semantic channel prep: s[b,n] scatter of top-10 cosine sims ----
    k_aspect<<<BB, NN, 0, stream>>>(h, amask, arepr, ainv);
    k_sim<<<ROWS / 8, 256, 0, stream>>>(h, arepr, ainv, sim);
    k_topk<<<BB, NN, 0, stream>>>(sim, sbuf);

    // ---- f16 staging ----
    long th = (long)ROWS * DD;
    k_f32_to_f16_pad<<<(unsigned)((th + 255) / 256), 256, 0, stream>>>(h, h16, DD, DD, th);
    auto convW = [&](const float* src, _Float16* dst, int K, int KPd) {
        long t = (long)GP * KPd;
        k_weightT<<<(unsigned)((t + 255) / 256), 256, 0, stream>>>(src, dst, K, GG, KPd);
    };
    convW(syn0_W,  w_s0,  DD, DD);
    convW(syn0_tW, w_s0t, DD, DD);
    convW(sem0_W,  w_m0,  DD, DD);
    convW(sem0_tW, w_m0t, DD, DD);
    convW(syn1_W,  w_s1,  GG, KP1);
    convW(sem1_W,  w_m1,  GG, KP1);
    convW(fus_W,   w_fus, 2 * GG, KPF);

    // ---- syntactic channel ----
    run_layer(h16, DD, w_s0, w_s0t, syn0_a1, syn0_a2, syn0_pos, syn0_tb, syn0_g, syn0_b,
              positions, syn_adj, nullptr, nullptr,
              Wh, res, hp, attn16, whT, f1, f2, hs0c, hs0_16, KP1, 0, stream);
    run_layer(hs0_16, KP1, w_s1, nullptr, syn1_a1, syn1_a2, syn1_pos, nullptr, syn1_g, syn1_b,
              positions, syn_adj, nullptr, hs0c,
              Wh, res, hp, attn16, whT, f1, f2, nullptr, fusA16, KPF, 0, stream);

    // ---- semantic channel ----
    run_layer(h16, DD, w_m0, w_m0t, sem0_a1, sem0_a2, sem0_pos, sem0_tb, sem0_g, sem0_b,
              positions, nullptr, sbuf, nullptr,
              Wh, res, hp, attn16, whT, f1, f2, hm0c, hm0_16, KP1, 0, stream);
    run_layer(hm0_16, KP1, w_m1, nullptr, sem1_a1, sem1_a2, sem1_pos, nullptr, sem1_g, sem1_b,
              positions, nullptr, sbuf, hm0c,
              Wh, res, hp, attn16, whT, f1, f2, nullptr, fusA16, KPF, GG, stream);

    // ---- fusion ----
    k_gemm<<<dim3(GP / 16, ROWS / 256, 1), 128, 0, stream>>>(
        fusA16, w_fus, Wh, ROWS, GP, KPF, KPF, KPF, GP, 0, 0, 0);
    long to = (long)ROWS * GG;
    k_bias_relu<<<(unsigned)((to + 255) / 256), 256, 0, stream>>>(Wh, fus_b, out);

    (void)in_sizes; (void)n_in; (void)out_size; (void)ws_size;
}
